// Net_31576599560688
// MI455X (gfx1250) — compile-verified, matrix operand order checked by reading, not run on hardware
//
#include <hip/hip_runtime.h>
#include <hip/hip_bf16.h>

// ---------------------------------------------------------------------------
// Problem constants (match reference)
// ---------------------------------------------------------------------------
#define NN    50000
#define EE    1600000
#define DIN   128
#define HH    4
#define DD1   100
#define DD2   20
#define BB    64

#define MPAD  50048          // NN rounded up to 64 (4 M-tiles per wave)
#define K1P   128            // DIN already a multiple of 32
#define N1P   112            // DD1 -> 7 N-tiles
#define K2P   416            // 400 rounded up to 32
#define N2P   32             // DD2 -> 2 N-tiles

typedef __attribute__((ext_vector_type(16))) __bf16  v16bf;
typedef __attribute__((ext_vector_type(8)))  __bf16  v8bf;
typedef __attribute__((ext_vector_type(8)))  float   v8f;

// fp32 -> bf16 RNE (used only in one-time pack kernels, off the hot path)
__device__ __forceinline__ __bf16 f2bf(float f) {
    unsigned int u = __float_as_uint(f);
    unsigned int r = (u + 0x7FFFu + ((u >> 16) & 1u)) >> 16;
    unsigned short s = (unsigned short)r;
    return __builtin_bit_cast(__bf16, s);
}

// ordered-int float atomic max (memory must be initialized to -inf)
__device__ __forceinline__ void atomicMaxF(float* addr, float val) {
    if (val >= 0.0f) {
        atomicMax((int*)addr, __float_as_int(val));
    } else {
        atomicMin((unsigned int*)addr, __float_as_uint(val));
    }
}

// ---------------------------------------------------------------------------
// Pack kernels: fp32 -> padded bf16 (A row-major; B transposed to [Npad x Kpad])
// ---------------------------------------------------------------------------
__global__ void pack_rows_bf16(const float* __restrict__ A, __bf16* __restrict__ out,
                               int M, int K, long Mpad, int Kpad)
{
    long i = (long)blockIdx.x * blockDim.x + threadIdx.x;
    if (i >= Mpad * Kpad) return;
    const int  c = (int)(i % Kpad);
    const long r = i / Kpad;
    const float v = (r < M && c < K) ? A[r * K + c] : 0.0f;
    out[i] = f2bf(v);
}

__global__ void pack_wt_bf16(const float* __restrict__ W, __bf16* __restrict__ out,
                             int K, int Nc, int Kpad, int Npad)
{
    int i = blockIdx.x * blockDim.x + threadIdx.x;
    if (i >= Npad * Kpad) return;
    const int k = i % Kpad, n = i / Kpad;
    const float v = (k < K && n < Nc) ? W[(long)k * Nc + n] : 0.0f;
    out[i] = f2bf(v);
}

// ---------------------------------------------------------------------------
// WMMA GEMM: C[M x Nc] = A[Mpad x Kpad](bf16) * Bt[Npad x Kpad](bf16, K-major)
// One wave owns a 64(M) x 16(N) strip: 4 accumulators share one B fragment per
// k-step -> hides WMMA->WMMA RAW latency. Zero bounds checks in the hot loop
// (padding guarantees validity); fragment halves load as single b128 each.
// Fragment layouts per CDNA5 ISA 7.12.2 (wave32).
// ---------------------------------------------------------------------------
__global__ __launch_bounds__(256)
void gemm_bf16_wmma(const __bf16* __restrict__ A, const __bf16* __restrict__ Bt,
                    float* __restrict__ C, int M, long Mpad, int Kpad, int Nc, int Npad)
{
    const int lane = threadIdx.x & 31;
    const int wave = threadIdx.x >> 5;
    const int tilesN  = Npad >> 4;
    const int groupsM = (int)(Mpad >> 6);
    const long wid = (long)blockIdx.x * 8 + wave;
    if (wid >= (long)groupsM * tilesN) return;
    const int gm = (int)(wid / tilesN);
    const int tn = (int)(wid % tilesN);

    const int half = lane >> 4;       // lane half selects K sub-range
    const int l16  = lane & 15;
    const long row0 = (long)gm * 64 + l16;      // A fragment rows (+16 per tile)
    const __bf16* bptr = Bt + ((long)(tn * 16 + l16)) * Kpad;

    v8f acc[4] = {v8f{}, v8f{}, v8f{}, v8f{}};

    for (int k0 = 0; k0 < Kpad; k0 += 32) {
        const int kA = k0 + half * 8;
        // B fragment: 2 x b128
        const v8bf b_lo = *(const v8bf*)(bptr + kA);
        const v8bf b_hi = *(const v8bf*)(bptr + kA + 16);
        const v16bf bfrag = __builtin_shufflevector(b_lo, b_hi,
            0,1,2,3,4,5,6,7,8,9,10,11,12,13,14,15);
        #pragma unroll
        for (int t = 0; t < 4; ++t) {
            const __bf16* aptr = A + (row0 + t * 16) * Kpad + kA;
            const v8bf a_lo = *(const v8bf*)(aptr);
            const v8bf a_hi = *(const v8bf*)(aptr + 16);
            const v16bf afrag = __builtin_shufflevector(a_lo, a_hi,
                0,1,2,3,4,5,6,7,8,9,10,11,12,13,14,15);
            acc[t] = __builtin_amdgcn_wmma_f32_16x16x32_bf16(
                false, afrag, false, bfrag, (short)0, acc[t], false, false);
        }
    }

    // C/D layout: VGPR r -> row base + r (+8 for lanes 16..31), col tn*16 + lane%16
    const int cCol = tn * 16 + l16;
    if (cCol < Nc) {
        #pragma unroll
        for (int t = 0; t < 4; ++t) {
            const long rbase = (long)gm * 64 + t * 16 + half * 8;
            #pragma unroll
            for (int r = 0; r < 8; ++r) {
                const long cRow = rbase + r;
                if (cRow < M) C[cRow * Nc + cCol] = acc[t][r];
            }
        }
    }
}

// ---------------------------------------------------------------------------
// Utility kernels
// ---------------------------------------------------------------------------
__global__ void fill_f32(float* __restrict__ p, float v, long n) {
    long i = (long)blockIdx.x * blockDim.x + threadIdx.x;
    if (i < n) p[i] = v;
}

__global__ void scale_relu_f32(float* __restrict__ p, float scale, long n) {
    long i = (long)blockIdx.x * blockDim.x + threadIdx.x;
    if (i < n) p[i] = fmaxf(p[i] * scale, 0.0f);
}

// per-(head,node) attention scores: s_src = z . a[:d], s_dst = z . a[d:]
__global__ void node_scores(const float* __restrict__ Z, const float* __restrict__ a,
                            float* __restrict__ ssrc, float* __restrict__ sdst,
                            int n, int d, int h)
{
    int idx = blockIdx.x * blockDim.x + threadIdx.x;   // (head, node)
    if (idx >= n * h) return;
    const int hd = idx / n, nd = idx % n;
    const float* z  = Z + ((long)hd * n + nd) * d;
    const float* av = a + (long)hd * 2 * d;
    float s0 = 0.0f, s1 = 0.0f;
    for (int c = 0; c < d; ++c) {
        const float zv = z[c];
        s0 += zv * av[c];
        s1 += zv * av[d + c];
    }
    ssrc[idx] = s0;
    sdst[idx] = s1;
}

__device__ __forceinline__ float leaky(float x) { return x > 0.0f ? x : 0.01f * x; }

// pass 1: per-destination segment max of leaky_relu(s_src[src] + s_dst[dst])
__global__ void edge_seg_max(const int* __restrict__ src, const int* __restrict__ dst,
                             const float* __restrict__ ssrc, const float* __restrict__ sdst,
                             float* __restrict__ m, int e, int n, int h)
{
    long idx = (long)blockIdx.x * blockDim.x + threadIdx.x;   // (head, edge)
    if (idx >= (long)e * h) return;
    const int hd = (int)(idx / e);
    const long ed = idx % e;
    const int s = src[ed], t = dst[ed];
    const float v = leaky(ssrc[hd * n + s] + sdst[hd * n + t]);
    atomicMaxF(&m[hd * n + t], v);
}

// pass 2: per-destination softmax denominator (recompute logit; inputs L2-resident)
__global__ void edge_seg_den(const int* __restrict__ src, const int* __restrict__ dst,
                             const float* __restrict__ ssrc, const float* __restrict__ sdst,
                             const float* __restrict__ m, float* __restrict__ den,
                             int e, int n, int h)
{
    long idx = (long)blockIdx.x * blockDim.x + threadIdx.x;
    if (idx >= (long)e * h) return;
    const int hd = (int)(idx / e);
    const long ed = idx % e;
    const int s = src[ed], t = dst[ed];
    const float v = leaky(ssrc[hd * n + s] + sdst[hd * n + t]);
    atomicAdd(&den[hd * n + t], __expf(v - m[hd * n + t]));
}

// pass 3: scatter alpha * z[src] into destination rows. One wave per (head,edge).
// Each active lane moves a float4 (one b128 load + 4 f32 atomics). d % 4 == 0.
// colStride = d -> concat heads (layer 1); colStride = 0 -> sum heads (layer 2).
__global__ __launch_bounds__(256)
void edge_scatter(const int* __restrict__ src, const int* __restrict__ dst,
                  const float* __restrict__ ssrc, const float* __restrict__ sdst,
                  const float* __restrict__ m, const float* __restrict__ den,
                  const float* __restrict__ Z, float* __restrict__ out,
                  int e, int n, int h, int d, int ldo, int colStride)
{
    const long wid = ((long)blockIdx.x * blockDim.x + threadIdx.x) >> 5;
    const int lane = threadIdx.x & 31;
    if (wid >= (long)e * h) return;
    const int hd = (int)(wid / e);
    const long ed = wid % e;
    const int s = src[ed], t = dst[ed];
    const float v = leaky(ssrc[hd * n + s] + sdst[hd * n + t]);
    const float alpha = __expf(v - m[hd * n + t]) / den[hd * n + t];
    const int c0 = lane * 4;
    if (c0 < d) {
        const float4 zv = *(const float4*)(Z + ((long)hd * n + s) * d + c0);
        float* orow = out + (long)t * ldo + (long)hd * colStride + c0;
        atomicAdd(orow + 0, alpha * zv.x);
        atomicAdd(orow + 1, alpha * zv.y);
        atomicAdd(orow + 2, alpha * zv.z);
        atomicAdd(orow + 3, alpha * zv.w);
    }
}

// per-graph feature sums + node counts
__global__ void graph_accum(const float* __restrict__ h2, const int* __restrict__ gid,
                            float* __restrict__ hg, float* __restrict__ cnt, int n, int d)
{
    int i = blockIdx.x * blockDim.x + threadIdx.x;
    if (i >= n) return;
    const int g = gid[i];
    atomicAdd(&cnt[g], 1.0f);
    const float* row = h2 + (long)i * d;
    for (int c = 0; c < d; ++c) atomicAdd(&hg[g * d + c], row[c]);
}

// whole readout MLP + BatchNorm in one block (B=64 rows)
__global__ __launch_bounds__(256)
void readout_mlp(const float* __restrict__ hg, const float* __restrict__ cnt,
                 const float* __restrict__ Wf1, const float* __restrict__ bf1,
                 const float* __restrict__ Wf2, const float* __restrict__ bf2,
                 const float* __restrict__ Wf3, const float* __restrict__ bf3,
                 const float* __restrict__ gamma, const float* __restrict__ beta,
                 float* __restrict__ out)
{
    __shared__ float x0[BB * DD2];
    __shared__ float x1[BB * 128];
    __shared__ float x2[BB * 32];
    __shared__ float mu[32], var[32];
    const int t = threadIdx.x;

    for (int i = t; i < BB * DD2; i += 256) {
        const int b = i / DD2;
        x0[i] = hg[i] / fmaxf(cnt[b], 1.0f);        // mean_nodes
    }
    __syncthreads();
    for (int i = t; i < BB * 128; i += 256) {       // 20 -> 128, relu
        const int b = i / 128, c = i % 128;
        float s = bf1[c];
        for (int k = 0; k < DD2; ++k) s += x0[b * DD2 + k] * Wf1[k * 128 + c];
        x1[i] = fmaxf(s, 0.0f);
    }
    __syncthreads();
    for (int i = t; i < BB * 32; i += 256) {        // 128 -> 32
        const int b = i / 32, c = i % 32;
        float s = bf2[c];
        for (int k = 0; k < 128; ++k) s += x1[b * 128 + k] * Wf2[k * 32 + c];
        x2[i] = s;
    }
    __syncthreads();
    if (t < 32) {                                   // BatchNorm stats over B
        float s = 0.0f;
        for (int b = 0; b < BB; ++b) s += x2[b * 32 + t];
        const float mm = s / (float)BB;
        float v = 0.0f;
        for (int b = 0; b < BB; ++b) { const float dd = x2[b * 32 + t] - mm; v += dd * dd; }
        mu[t] = mm; var[t] = v / (float)BB;
    }
    __syncthreads();
    for (int i = t; i < BB * 32; i += 256) {        // normalize + relu
        const int c = i % 32;
        const float xn = gamma[c] * (x2[i] - mu[c]) * rsqrtf(var[c] + 1e-5f) + beta[c];
        x2[i] = fmaxf(xn, 0.0f);
    }
    __syncthreads();
    if (t < BB) {                                   // 32 -> 1
        float s = bf3[0];
        for (int k = 0; k < 32; ++k) s += x2[t * 32 + k] * Wf3[k];
        out[t] = s;
    }
}

// ---------------------------------------------------------------------------
// Host launch
// ---------------------------------------------------------------------------
static inline int cdiv(long a, long b) { return (int)((a + b - 1) / b); }

extern "C" void kernel_launch(void* const* d_in, const int* in_sizes, int n_in,
                              void* d_out, int out_size, void* d_ws, size_t ws_size,
                              hipStream_t stream)
{
    const float* feat = (const float*)d_in[0];
    const int*   src  = (const int*)  d_in[1];
    const int*   dst  = (const int*)  d_in[2];
    const int*   gid  = (const int*)  d_in[3];
    const float* W1   = (const float*)d_in[4];
    const float* a1   = (const float*)d_in[5];
    const float* W2   = (const float*)d_in[6];
    const float* a2   = (const float*)d_in[7];
    const float* Wf1  = (const float*)d_in[8];
    const float* bf1  = (const float*)d_in[9];
    const float* Wf2  = (const float*)d_in[10];
    const float* bf2  = (const float*)d_in[11];
    const float* Wf3  = (const float*)d_in[12];
    const float* bf3  = (const float*)d_in[13];
    const float* gam  = (const float*)d_in[14];
    const float* bet  = (const float*)d_in[15];
    float* outp = (float*)d_out;

    // workspace layout (64B aligned chunks; z2 reuses z1 after layer 1)
    float* ws  = (float*)d_ws;
    float* z1  = ws;                                  // H*N*D1 fp32
    float* h1  = z1 + (size_t)HH * NN * DD1;          // N*400 fp32
    float* h2  = h1 + (size_t)NN * HH * DD1;          // N*20 fp32
    float* ss  = h2 + (size_t)NN * DD2;               // H*N
    float* sd  = ss + (size_t)HH * NN;                // H*N
    float* mb  = sd + (size_t)HH * NN;                // H*N
    float* dn  = mb + (size_t)HH * NN;                // H*N
    float* hg  = dn + (size_t)HH * NN;                // B*D2
    float* cnt = hg + (size_t)BB * DD2;               // B
    __bf16* featbf = (__bf16*)(cnt + 64);             // MPAD*K1P bf16
    __bf16* h1bf   = featbf + (size_t)MPAD * K1P;     // MPAD*K2P bf16
    __bf16* w1t    = h1bf + (size_t)MPAD * K2P;       // H*N1P*K1P bf16
    __bf16* w2t    = w1t + (size_t)HH * N1P * K1P;    // H*N2P*K2P bf16
    float*  z2     = z1;

    const float ninf = -__builtin_inff();
    const long nh = (long)HH * NN;
    const long eh = (long)EE * HH;

    // ----- pack inputs to padded bf16 -----
    pack_rows_bf16<<<cdiv((long)MPAD * K1P, 256), 256, 0, stream>>>(
        feat, featbf, NN, DIN, MPAD, K1P);
    for (int h = 0; h < HH; ++h) {
        pack_wt_bf16<<<cdiv((long)N1P * K1P, 256), 256, 0, stream>>>(
            W1 + (size_t)h * DIN * DD1, w1t + (size_t)h * N1P * K1P, DIN, DD1, K1P, N1P);
        pack_wt_bf16<<<cdiv((long)N2P * K2P, 256), 256, 0, stream>>>(
            W2 + (size_t)h * (HH * DD1) * DD2, w2t + (size_t)h * N2P * K2P,
            HH * DD1, DD2, K2P, N2P);
    }

    // ----- layer 1 -----
    fill_f32<<<cdiv((long)NN * HH * DD1, 256), 256, 0, stream>>>(h1, 0.0f, (long)NN * HH * DD1);
    fill_f32<<<cdiv(nh, 256), 256, 0, stream>>>(mb, ninf, nh);
    fill_f32<<<cdiv(nh, 256), 256, 0, stream>>>(dn, 0.0f, nh);

    {
        const long waves = (long)(MPAD / 64) * (N1P / 16);
        for (int h = 0; h < HH; ++h)
            gemm_bf16_wmma<<<cdiv(waves, 8), 256, 0, stream>>>(
                featbf, w1t + (size_t)h * N1P * K1P, z1 + (size_t)h * NN * DD1,
                NN, MPAD, K1P, DD1, N1P);
    }
    node_scores<<<cdiv(nh, 256), 256, 0, stream>>>(z1, a1, ss, sd, NN, DD1, HH);
    edge_seg_max<<<cdiv(eh, 256), 256, 0, stream>>>(src, dst, ss, sd, mb, EE, NN, HH);
    edge_seg_den<<<cdiv(eh, 256), 256, 0, stream>>>(src, dst, ss, sd, mb, dn, EE, NN, HH);
    edge_scatter<<<cdiv(eh * 32, 256), 256, 0, stream>>>(
        src, dst, ss, sd, mb, dn, z1, h1, EE, NN, HH, DD1, HH * DD1, DD1);
    scale_relu_f32<<<cdiv((long)NN * HH * DD1, 256), 256, 0, stream>>>(
        h1, 1.0f, (long)NN * HH * DD1);

    // ----- layer 2 -----
    pack_rows_bf16<<<cdiv((long)MPAD * K2P, 256), 256, 0, stream>>>(
        h1, h1bf, NN, HH * DD1, MPAD, K2P);
    fill_f32<<<cdiv((long)NN * DD2, 256), 256, 0, stream>>>(h2, 0.0f, (long)NN * DD2);
    fill_f32<<<cdiv(nh, 256), 256, 0, stream>>>(mb, ninf, nh);
    fill_f32<<<cdiv(nh, 256), 256, 0, stream>>>(dn, 0.0f, nh);

    {
        const long waves = (long)(MPAD / 64) * (N2P / 16);
        for (int h = 0; h < HH; ++h)
            gemm_bf16_wmma<<<cdiv(waves, 8), 256, 0, stream>>>(
                h1bf, w2t + (size_t)h * N2P * K2P, z2 + (size_t)h * NN * DD2,
                NN, MPAD, K2P, DD2, N2P);
    }
    node_scores<<<cdiv(nh, 256), 256, 0, stream>>>(z2, a2, ss, sd, NN, DD2, HH);
    edge_seg_max<<<cdiv(eh, 256), 256, 0, stream>>>(src, dst, ss, sd, mb, EE, NN, HH);
    edge_seg_den<<<cdiv(eh, 256), 256, 0, stream>>>(src, dst, ss, sd, mb, dn, EE, NN, HH);
    edge_scatter<<<cdiv(eh * 32, 256), 256, 0, stream>>>(
        src, dst, ss, sd, mb, dn, z2, h2, EE, NN, HH, DD2, DD2, 0);
    scale_relu_f32<<<cdiv((long)NN * DD2, 256), 256, 0, stream>>>(
        h2, 1.0f / HH, (long)NN * DD2);

    // ----- readout -----
    fill_f32<<<1, 256, 0, stream>>>(hg, 0.0f, (long)BB * DD2);
    fill_f32<<<1, 64, 0, stream>>>(cnt, 0.0f, (long)BB);
    graph_accum<<<cdiv(NN, 256), 256, 0, stream>>>(h2, gid, hg, cnt, NN, DD2);
    readout_mlp<<<1, 256, 0, stream>>>(hg, cnt, Wf1, bf1, Wf2, bf2, Wf3, bf3, gam, bet, outp);

    (void)in_sizes; (void)n_in; (void)out_size; (void)ws_size;
}
